// Regress_onset_offset_frame_velocity_S4_68779606278803
// MI455X (gfx1250) — matrix operand order repaired and checked by reading, not compiled, not run on hardware
//
#include <hip/hip_runtime.h>
#include <hip/hip_bf16.h>
#include <math.h>

// ---------------- problem constants ----------------
#define B_   2
#define L_   160000
#define H_   128
#define N_   64
#define NL_  4
#define TS_  1001
#define MR_  16
#define CLS_ 88
#define LMAX_ 1024
#define REP_ 16016

#define PAD_   1040                 // zero prefix so conv never branches
#define LSTR_  (L_ + PAD_)          // 161040 halfs per (b,h) row
#define NTILES_ 625                 // L/256
#define TG_     5                   // tiles per wave-group (5 accumulators in flight)
#define NGRP_   (NTILES_ / TG_)     // 125 groups per (b,h)
#define NDI_    33                  // block-diagonal steps (d = 0,2,...,64)

typedef _Float16 half_t;
typedef __attribute__((ext_vector_type(16))) _Float16 v16h;
typedef __attribute__((ext_vector_type(8)))  float    v8f;

struct alignas(16) h8 { half_t v[8]; };

__device__ __forceinline__ v16h load_frag(const half_t* p0, const half_t* p1) {
    h8 a = *(const h8*)p0;
    h8 b = *(const h8*)p1;
    v16h f;
#pragma unroll
    for (int j = 0; j < 8; ++j) { f[j] = a.v[j]; f[8 + j] = b.v[j]; }
    return f;
}

// branchless fast activations (single v_exp_f32 each, saturate correctly)
__device__ __forceinline__ float sigmoid_fast(float x) {
    return 1.0f / (1.0f + __expf(-x));
}
__device__ __forceinline__ float gelu_fast(float x) {
    float z = 0.7978845608028654f * (x + 0.044715f * x * x * x);
    float t = 1.0f - 2.0f / (1.0f + __expf(2.0f * z));   // tanh(z)
    return 0.5f * x * (1.0f + t);
}

// ---------------- tiny prep kernels ----------------

// out_w (NL,2H,H) f32 -> f16 copy
__global__ void wf16_kernel(const float* __restrict__ w, half_t* __restrict__ wf) {
    int idx = blockIdx.x * blockDim.x + threadIdx.x;
    if (idx < NL_ * 2 * H_ * H_) wf[idx] = (half_t)w[idx];
}

// encoder: u0[b][h][PAD+l] = f16(x[b,l]*enc_w[h]+enc_b[h]); pad region zeroed
__global__ void encode_kernel(const float* __restrict__ x, const float* __restrict__ ew,
                              const float* __restrict__ eb, half_t* __restrict__ u0) {
    size_t total = (size_t)B_ * H_ * LSTR_;
    for (size_t idx = (size_t)blockIdx.x * blockDim.x + threadIdx.x; idx < total;
         idx += (size_t)gridDim.x * blockDim.x) {
        int i = (int)(idx % LSTR_);
        size_t bh = idx / LSTR_;
        int h = (int)(bh % H_);
        int b = (int)(bh / H_);
        half_t v;
        if (i < PAD_) v = (half_t)0.0f;
        else {
            int l = i - PAD_;
            v = (half_t)(x[(size_t)b * L_ + l] * ew[h] + eb[h]);
        }
        u0[idx] = v;
    }
}

// zero the pad prefix of the second ping-pong buffer
__global__ void zero_pad_kernel(half_t* __restrict__ u) {
    int idx = blockIdx.x * blockDim.x + threadIdx.x;
    if (idx >= B_ * H_ * PAD_) return;
    int i = idx % PAD_;
    int bh = idx / PAD_;
    u[(size_t)bh * LSTR_ + i] = (half_t)0.0f;
}

// SSM kernel K[h,l] = 2*Re( sum_n Ct[h,n] * exp(dtA[h,n]*l) ), layer slice pointers
__global__ void kraw_kernel(const float* __restrict__ log_dt, const float* __restrict__ lar,
                            const float* __restrict__ aim, const float* __restrict__ cre,
                            const float* __restrict__ cim, float* __restrict__ Kraw) {
    int idx = blockIdx.x * blockDim.x + threadIdx.x;
    if (idx >= H_ * LMAX_) return;
    int h = idx / LMAX_;
    int l = idx % LMAX_;
    float dt = expf(log_dt[h]);
    float acc = 0.0f;
    for (int n = 0; n < N_; ++n) {
        float Are = -expf(lar[h * N_ + n]);
        float Aim = aim[h * N_ + n];
        float dre = Are * dt, dim_ = Aim * dt;
        float er = expf(dre);
        float sn, cs;
        __sincosf(dim_, &sn, &cs);
        float Ere = er * cs - 1.0f, Eim = er * sn;       // exp(dtA)-1
        float cr = cre[h * N_ + n], ci = cim[h * N_ + n];
        float numr = cr * Ere - ci * Eim;
        float numi = cr * Eim + ci * Ere;
        float den = Are * Are + Aim * Aim;
        float ctr = (numr * Are + numi * Aim) / den;      // Ct = C*(e^{dtA}-1)/A
        float cti = (numi * Are - numr * Aim) / den;
        float ve = expf(dre * (float)l);
        float vs, vc;
        __sincosf(dim_ * (float)l, &vs, &vc);
        acc += ctr * (ve * vc) - cti * (ve * vs);
    }
    Kraw[idx] = 2.0f * acc;
}

// Toeplitz A-tiles: KT[h][di][m][k] = Kraw[h, 32*di + 16 + m - k] (0 outside [0,LMAX))
__global__ void kt_kernel(const float* __restrict__ Kraw, half_t* __restrict__ KT) {
    int idx = blockIdx.x * blockDim.x + threadIdx.x;
    if (idx >= H_ * NDI_ * 512) return;
    int k = idx & 31;
    int m = (idx >> 5) & 15;
    int rest = idx >> 9;
    int di = rest % NDI_;
    int h = rest / NDI_;
    int t = 32 * di + 16 + m - k;
    float v = (t >= 0 && t < LMAX_) ? Kraw[h * LMAX_ + t] : 0.0f;
    KT[idx] = (half_t)v;
}

// ---------------- WMMA depthwise conv (block-Toeplitz) + D-skip + GELU ----------------
// one wave per block; block -> (b, h, group of TG_ tiles of 256 outputs).
// di is the OUTER loop: one A (Toeplitz) fragment feeds TG_ independent WMMAs,
// so the scheduler can overlap B-tile loads with matrix ops.
__global__ __launch_bounds__(32)
void conv_kernel(const half_t* __restrict__ u, const half_t* __restrict__ KT,
                 const float* __restrict__ Dvec, half_t* __restrict__ y) {
    int blk = blockIdx.x;
    int grp = blk % NGRP_;
    int h = (blk / NGRP_) % H_;
    int b = blk / (NGRP_ * H_);
    int lane = threadIdx.x;
    int m16 = lane & 15;
    int hi = lane >> 4;
    int kbA = hi ? 8 : 0;    // A-frag K base per ISA A-layout
    int kbB = hi ? 8 : 0;    // B-frag K base

    const half_t* urow = u + ((size_t)(b * H_ + h)) * LSTR_ + PAD_;
    const half_t* ktb = KT + (size_t)h * (NDI_ * 512);
    half_t* yrow = y + ((size_t)(b * H_ + h)) * L_;
    float Dh = Dvec[h];

    int P0 = grp * (TG_ * 256);
    v8f acc[TG_];
#pragma unroll
    for (int g = 0; g < TG_; ++g)
        acc[g] = (v8f){0.f, 0.f, 0.f, 0.f, 0.f, 0.f, 0.f, 0.f};

#pragma unroll 1
    for (int di = 0; di < NDI_; ++di) {
        const half_t* ap = ktb + di * 512 + m16 * 32 + kbA;
        v16h a = load_frag(ap, ap + 16);
        const half_t* bbase = urow + (P0 + 16 * (m16 - 2 * di) - 16 + kbB);
#pragma unroll
        for (int g = 0; g < TG_; ++g) {
            const half_t* bp = bbase + g * 256;
            v16h bb = load_frag(bp, bp + 16);
            acc[g] = __builtin_amdgcn_wmma_f32_16x16x32_f16(false, a, false, bb,
                                                            (short)0, acc[g], false, false);
        }
    }

    // D[m,n] = y[P0 + 16n + m]; n = lane&15, m = r + 8*hi -> 8 consecutive positions
#pragma unroll
    for (int g = 0; g < TG_; ++g) {
        int pbase = P0 + g * 256 + 16 * m16 + 8 * hi;
        h8 uv = *(const h8*)(urow + pbase);
        h8 out;
#pragma unroll
        for (int r = 0; r < 8; ++r) {
            float v = acc[g][r] + Dh * (float)uv.v[r];
            out.v[r] = (half_t)gelu_fast(v);
        }
        *(h8*)(yrow + pbase) = out;
    }
}

// ---------------- fused GLU linear (WMMA) + residual + LayerNorm ----------------
// block = 128 threads (4 waves) handling 16 positions, full 2H x H GEMM
__global__ __launch_bounds__(128)
void glu_ln_kernel(const half_t* __restrict__ y, const half_t* __restrict__ ucur,
                   const half_t* __restrict__ W, const float* __restrict__ ob,
                   const float* __restrict__ lng, const float* __restrict__ lnb,
                   half_t* __restrict__ unext) {
    __shared__ alignas(16) half_t yT[16 * 136];  // transposed y tile [p][h], stride 136
    __shared__ alignas(16) half_t uT[128 * 16];  // u tile [h][p]
    __shared__ float RES[128 * 16];              // glu+residual, [h][p]
    __shared__ float MU[16], RS[16];

    int blk = blockIdx.x;
    int pt = blk % (L_ / 16);
    int b = blk / (L_ / 16);
    int p0 = pt * 16;
    int t = threadIdx.x;

    {   // stage: thread t owns channel row h=t
        const half_t* yr = y + ((size_t)(b * H_ + t)) * L_ + p0;
        h8 y0 = *(const h8*)yr;
        h8 y1 = *(const h8*)(yr + 8);
#pragma unroll
        for (int j = 0; j < 8; ++j) {
            yT[j * 136 + t] = y0.v[j];
            yT[(j + 8) * 136 + t] = y1.v[j];
        }
        const half_t* ur = ucur + ((size_t)(b * H_ + t)) * LSTR_ + PAD_ + p0;
        *(h8*)(uT + t * 16) = *(const h8*)ur;
        *(h8*)(uT + t * 16 + 8) = *(const h8*)(ur + 8);
    }
    __syncthreads();

    int lane = t & 31;
    int w = t >> 5;
    int m16 = lane & 15;
    int hi = lane >> 4;
    int kbA = hi ? 8 : 0;
    int kbB = hi ? 8 : 0;

#pragma unroll
    for (int half_idx = 0; half_idx < 2; ++half_idx) {
        int mtA = w + 4 * half_idx;   // rows [16*mtA, 16*mtA+16) of z (h < 128)
        int mtB = mtA + 8;            // gate rows (+128)
        v8f accA = {0.f, 0.f, 0.f, 0.f, 0.f, 0.f, 0.f, 0.f};
        v8f accB = {0.f, 0.f, 0.f, 0.f, 0.f, 0.f, 0.f, 0.f};
#pragma unroll
        for (int kc = 0; kc < 4; ++kc) {
            int k0 = kc * 32;
            const half_t* bp = &yT[m16 * 136 + k0 + kbB];
            v16h bb = load_frag(bp, bp + 16);                // ds_load_b128 x2
            const half_t* apA = W + (size_t)(mtA * 16 + m16) * H_ + k0 + kbA;
            v16h aA = load_frag(apA, apA + 16);
            accA = __builtin_amdgcn_wmma_f32_16x16x32_f16(false, aA, false, bb,
                                                          (short)0, accA, false, false);
            const half_t* apB = W + (size_t)(mtB * 16 + m16) * H_ + k0 + kbA;
            v16h aB = load_frag(apB, apB + 16);
            accB = __builtin_amdgcn_wmma_f32_16x16x32_f16(false, aB, false, bb,
                                                          (short)0, accB, false, false);
        }
#pragma unroll
        for (int r = 0; r < 8; ++r) {
            int m = r + 8 * hi;
            int hA = mtA * 16 + m;
            float a = accA[r] + ob[hA];
            float g = accB[r] + ob[128 + hA];
            float glu = a * sigmoid_fast(g);
            RES[hA * 16 + m16] = glu + (float)uT[hA * 16 + m16];
        }
    }
    __syncthreads();

    if (t < 16) {   // LN stats per position
        float s = 0.f, s2 = 0.f;
        for (int hh = 0; hh < 128; ++hh) {
            float v = RES[hh * 16 + t];
            s += v; s2 += v * v;
        }
        float mu = s * (1.0f / 128.0f);
        float var = s2 * (1.0f / 128.0f) - mu * mu;
        MU[t] = mu;
        RS[t] = rsqrtf(var + 1e-5f);
    }
    __syncthreads();

    {   // normalized write-back, coalesced rows
        float g = lng[t], bb = lnb[t];
        half_t* ur = unext + ((size_t)(b * H_ + t)) * LSTR_ + PAD_ + p0;
        h8 o0, o1;
#pragma unroll
        for (int j = 0; j < 8; ++j) {
            o0.v[j] = (half_t)((RES[t * 16 + j] - MU[j]) * RS[j] * g + bb);
            o1.v[j] = (half_t)((RES[t * 16 + 8 + j] - MU[8 + j]) * RS[8 + j] * g + bb);
        }
        *(h8*)ur = o0;
        *(h8*)(ur + 8) = o1;
    }
}

// ---------------- tail kernels ----------------
__global__ __launch_bounds__(256)
void pool_kernel(const half_t* __restrict__ u, float* __restrict__ feat) {
    int bh = blockIdx.x;
    const half_t* ur = u + (size_t)bh * LSTR_ + PAD_;
    float s = 0.f;
    for (int i = threadIdx.x; i < L_; i += 256) s += (float)ur[i];
    __shared__ float red[256];
    red[threadIdx.x] = s;
    __syncthreads();
    for (int st = 128; st > 0; st >>= 1) {
        if (threadIdx.x < st) red[threadIdx.x] += red[threadIdx.x + st];
        __syncthreads();
    }
    if (threadIdx.x == 0) feat[bh] = red[0] / (float)L_;
}

__global__ void dec_kernel(const float* __restrict__ feat, const float* __restrict__ dw,
                           const float* __restrict__ db, float* __restrict__ rep) {
    int idx = blockIdx.x * blockDim.x + threadIdx.x;
    if (idx >= B_ * REP_) return;
    int b = idx / REP_, r = idx % REP_;
    float s = db[r];
    for (int h = 0; h < H_; ++h) s += feat[b * H_ + h] * dw[(size_t)h * REP_ + r];
    rep[idx] = s;
}

__global__ void head_kernel(const float* __restrict__ rep, const float* __restrict__ hw,
                            const float* __restrict__ hb, float* __restrict__ out) {
    int idx = blockIdx.x * blockDim.x + threadIdx.x;
    if (idx >= 4 * B_ * TS_ * CLS_) return;
    int c = idx % CLS_;
    int tmp = idx / CLS_;
    int tq = tmp % TS_;
    int b = (tmp / TS_) % B_;
    int k = tmp / (TS_ * B_);
    float s = hb[k * CLS_ + c];
#pragma unroll
    for (int m = 0; m < MR_; ++m)
        s += rep[(b * TS_ + tq) * MR_ + m] * hw[(k * MR_ + m) * CLS_ + c];
    out[idx] = s;
}

// ---------------- launch ----------------
static inline size_t align256(size_t x) { return (x + 255) & ~(size_t)255; }

extern "C" void kernel_launch(void* const* d_in, const int* in_sizes, int n_in,
                              void* d_out, int out_size, void* d_ws, size_t ws_size,
                              hipStream_t stream) {
    const float* x      = (const float*)d_in[0];
    const float* enc_w  = (const float*)d_in[1];
    const float* enc_b  = (const float*)d_in[2];
    const float* log_dt = (const float*)d_in[3];
    const float* lar    = (const float*)d_in[4];
    const float* aim    = (const float*)d_in[5];
    const float* cre    = (const float*)d_in[6];
    const float* cim    = (const float*)d_in[7];
    const float* Dv     = (const float*)d_in[8];
    const float* out_w  = (const float*)d_in[9];
    const float* out_b  = (const float*)d_in[10];
    const float* ln_g   = (const float*)d_in[11];
    const float* ln_b   = (const float*)d_in[12];
    const float* dec_w  = (const float*)d_in[13];
    const float* dec_b  = (const float*)d_in[14];
    const float* head_w = (const float*)d_in[15];
    const float* head_b = (const float*)d_in[16];
    float* out = (float*)d_out;

    // workspace carve-up
    char* ws = (char*)d_ws;
    size_t off = 0;
    const size_t SZ_U = (size_t)B_ * H_ * LSTR_ * sizeof(half_t);
    half_t* ubuf0 = (half_t*)(ws + off); off = align256(off + SZ_U);
    half_t* ubuf1 = (half_t*)(ws + off); off = align256(off + SZ_U);
    half_t* ybuf  = (half_t*)(ws + off); off = align256(off + (size_t)B_ * H_ * L_ * sizeof(half_t));
    float*  Kraw  = (float*)(ws + off);  off = align256(off + (size_t)H_ * LMAX_ * sizeof(float));
    half_t* KT    = (half_t*)(ws + off); off = align256(off + (size_t)H_ * NDI_ * 512 * sizeof(half_t));
    half_t* Wf16  = (half_t*)(ws + off); off = align256(off + (size_t)NL_ * 2 * H_ * H_ * sizeof(half_t));
    float*  feat  = (float*)(ws + off);  off = align256(off + (size_t)B_ * H_ * sizeof(float));
    float*  rep   = (float*)(ws + off);  off = align256(off + (size_t)B_ * REP_ * sizeof(float));
    (void)ws_size; (void)in_sizes; (void)n_in; (void)out_size;

    // prep
    wf16_kernel<<<(NL_ * 2 * H_ * H_ + 255) / 256, 256, 0, stream>>>(out_w, Wf16);
    encode_kernel<<<4096, 256, 0, stream>>>(x, enc_w, enc_b, ubuf0);
    zero_pad_kernel<<<(B_ * H_ * PAD_ + 255) / 256, 256, 0, stream>>>(ubuf1);

    half_t* ucur = ubuf0;
    half_t* unext = ubuf1;
    for (int layer = 0; layer < NL_; ++layer) {
        kraw_kernel<<<(H_ * LMAX_ + 255) / 256, 256, 0, stream>>>(
            log_dt + layer * H_, lar + (size_t)layer * H_ * N_, aim + (size_t)layer * H_ * N_,
            cre + (size_t)layer * H_ * N_, cim + (size_t)layer * H_ * N_, Kraw);
        kt_kernel<<<(H_ * NDI_ * 512 + 255) / 256, 256, 0, stream>>>(Kraw, KT);
        conv_kernel<<<B_ * H_ * NGRP_, 32, 0, stream>>>(ucur, KT, Dv + layer * H_, ybuf);
        glu_ln_kernel<<<B_ * (L_ / 16), 128, 0, stream>>>(
            ybuf, ucur, Wf16 + (size_t)layer * 2 * H_ * H_, out_b + layer * 2 * H_,
            ln_g + layer * H_, ln_b + layer * H_, unext);
        half_t* tmp = ucur; ucur = unext; unext = tmp;
    }

    pool_kernel<<<B_ * H_, 256, 0, stream>>>(ucur, feat);
    dec_kernel<<<(B_ * REP_ + 255) / 256, 256, 0, stream>>>(feat, dec_w, dec_b, rep);
    head_kernel<<<(4 * B_ * TS_ * CLS_ + 255) / 256, 256, 0, stream>>>(rep, head_w, head_b, out);
}
